// QKVAttention_8624294331206
// MI455X (gfx1250) — compile-verified, matrix operand order checked by reading, not run on hardware
//
#include <hip/hip_runtime.h>
#include <math.h>

// ---------------------------------------------------------------------------
// Problem constants (fixed by the reference)
// ---------------------------------------------------------------------------
#define MODEL_DIM 2048
#define NUM_HEADS 16
#define HEAD_DIM  128
#define BATCH     4
#define SEQ       2048
#define MROWS     (BATCH * SEQ)          // 8192

typedef __attribute__((ext_vector_type(16))) __bf16 v16bf;
typedef __attribute__((ext_vector_type(8)))  float  v8f;

// fp32 -> bf16 round-to-nearest-even
__device__ __forceinline__ unsigned short f2bf(float f) {
    union { float f; unsigned u; } x; x.f = f;
    unsigned u = x.u;
    unsigned r = (u + 0x7FFFu + ((u >> 16) & 1u)) >> 16;
    return (unsigned short)r;
}
__device__ __forceinline__ unsigned packbf(float a, float b) {
    return (unsigned)f2bf(a) | ((unsigned)f2bf(b) << 16);
}
__device__ __forceinline__ __bf16 bfu(unsigned short u) {
    return __builtin_bit_cast(__bf16, u);
}

// WMMA 16x16 bf16 operand K-index map (ISA 7.12.2, 16-bit A/B layout):
// lanes 0-15 : elems 0..7 -> K 0..7,  elems 8..15 -> K 16..23
// lanes 16-31: elems 0..7 -> K 8..15, elems 8..15 -> K 24..31
__device__ __forceinline__ int kmap(int e, int half) {
    return (e < 8 ? e : e + 8) + 8 * half;
}

// reduce across the 16-lane half-groups (rows of C-layout live across 16 lanes)
__device__ __forceinline__ float red_max16(float v) {
    v = fmaxf(v, __shfl_xor(v, 1, 32));
    v = fmaxf(v, __shfl_xor(v, 2, 32));
    v = fmaxf(v, __shfl_xor(v, 4, 32));
    v = fmaxf(v, __shfl_xor(v, 8, 32));
    return v;
}
__device__ __forceinline__ float red_sum16(float v) {
    v += __shfl_xor(v, 1, 32);
    v += __shfl_xor(v, 2, 32);
    v += __shfl_xor(v, 4, 32);
    v += __shfl_xor(v, 8, 32);
    return v;
}

// ---------------------------------------------------------------------------
// GEMM: C[M,N] = A[M,K] @ W[N,K]^T + bias[N]
//   mode 0: write fp32 [M,N]                      (final projection)
//   mode 1: write fp32 permuted [B,H,T,hd]        (K, V -> d_out)
//   mode 2: write bf16 permuted [B,H,T,hd]        (Q -> workspace)
// Block: 256 threads (8 waves). Tile 128(M) x 128(N), K step 32.
// ---------------------------------------------------------------------------
#define TM 128
#define TN 128
#define TK 32

__global__ __launch_bounds__(256) void gemm_bias_kernel(
    const float* __restrict__ A,
    const float* __restrict__ W,
    const float* __restrict__ bias,
    float* __restrict__ outF,
    unsigned short* __restrict__ outBF,
    int M, int N, int K, int mode)
{
    __shared__ unsigned short As[TM][TK + 2];
    __shared__ unsigned short Bs[TN][TK + 2];

    const int bm   = blockIdx.x * TM;
    const int bn   = blockIdx.y * TN;
    const int t    = threadIdx.x;
    const int wave = t >> 5;
    const int lane = t & 31;
    const int half = lane >> 4;
    const int l16  = lane & 15;

    v8f acc[8];
#pragma unroll
    for (int i = 0; i < 8; ++i) acc[i] = (v8f){0,0,0,0,0,0,0,0};

    for (int k0 = 0; k0 < K; k0 += TK) {
        // stage A and W tiles into LDS as bf16: float4 loads, packed dword stores
#pragma unroll
        for (int i = 0; i < 4; ++i) {
            int idx = t + 256 * i;            // 0..1023
            int r   = idx >> 3;               // 0..127
            int c   = (idx & 7) * 4;          // 0..28
            const float4 av = *(const float4*)(A + (size_t)(bm + r) * K + k0 + c);
            const float4 wv = *(const float4*)(W + (size_t)(bn + r) * K + k0 + c);
            *(unsigned*)&As[r][c]     = packbf(av.x, av.y);
            *(unsigned*)&As[r][c + 2] = packbf(av.z, av.w);
            *(unsigned*)&Bs[r][c]     = packbf(wv.x, wv.y);
            *(unsigned*)&Bs[r][c + 2] = packbf(wv.z, wv.w);
        }
        __syncthreads();

        v16bf afrag;
#pragma unroll
        for (int e = 0; e < 16; ++e)
            afrag[e] = bfu(As[wave * 16 + l16][kmap(e, half)]);

#pragma unroll
        for (int nt = 0; nt < 8; ++nt) {
            v16bf bfrag;
#pragma unroll
            for (int e = 0; e < 16; ++e)
                bfrag[e] = bfu(Bs[nt * 16 + l16][kmap(e, half)]);
            acc[nt] = __builtin_amdgcn_wmma_f32_16x16x32_bf16(
                false, afrag, false, bfrag, (short)0, acc[nt], false, false);
        }
        __syncthreads();
    }

    const int mbase = bm + wave * 16 + 8 * half;
#pragma unroll
    for (int nt = 0; nt < 8; ++nt) {
        int n = bn + nt * 16 + l16;
        float bv = bias[n];
#pragma unroll
        for (int e = 0; e < 8; ++e) {
            int r = mbase + e;
            float v = acc[nt][e] + bv;
            if (mode == 0) {
                outF[(size_t)r * N + n] = v;
            } else {
                int b = r >> 11, tt = r & (SEQ - 1);
                int h = n >> 7,  d  = n & (HEAD_DIM - 1);
                size_t o = (((size_t)(b * NUM_HEADS + h) * SEQ) + tt) * HEAD_DIM + d;
                if (mode == 1) outF[o] = v;
                else           outBF[o] = f2bf(v);
            }
        }
    }
}

// ---------------------------------------------------------------------------
// Causal flash attention.
// Grid: (B*H, T/64). Block: 128 threads = 4 waves; each wave owns 16 q rows.
// Q tile staged via GLOBAL_LOAD_ASYNC_TO_LDS_B128 (ASYNCcnt path).
// K/V tiles staged once per block into LDS as bf16 (shared by all 4 waves).
// ---------------------------------------------------------------------------
__global__ __launch_bounds__(128) void attn_kernel(
    const unsigned short* __restrict__ Qbf,  // [B,H,T,hd] bf16
    const float* __restrict__ Kf,            // [B,H,T,hd] fp32
    const float* __restrict__ Vf,            // [B,H,T,hd] fp32
    float* __restrict__ Oout)                // [B,T,H*hd] fp32
{
    __shared__ union SMem {
        unsigned short Qs[64][HEAD_DIM];                 // 16 KB (init phase only)
        struct {
            unsigned short Ks[32][HEAD_DIM + 2];
            unsigned short Vs[32][HEAD_DIM + 2];
            unsigned short Ps[4][16][34];
        } kv;                                            // ~21 KB (loop phase)
    } sm;

    const int bh   = blockIdx.x;             // b*H + h
    const int qblk = blockIdx.y;             // 64 q rows per block
    const int b    = bh >> 4;
    const int h    = bh & 15;
    const int t    = threadIdx.x;
    const int wave = t >> 5;
    const int lane = t & 31;
    const int half = lane >> 4;
    const int l16  = lane & 15;
    const int q0   = qblk * 64 + wave * 16;

    const size_t base = (size_t)bh * SEQ * HEAD_DIM;
    const float scale = 0.08838834764831845f;   // 1/sqrt(128)

    // ---- async-copy the 16KB Q tile (bf16, contiguous) into LDS ----
    {
        const unsigned short* qg = Qbf + base + (size_t)qblk * 64 * HEAD_DIM;
        unsigned lds0 = (unsigned)(uintptr_t)&sm.Qs[0][0];
        unsigned off  = (unsigned)t * 16u;
#pragma unroll
        for (int i = 0; i < 8; ++i) {
            unsigned la = lds0 + off + (unsigned)i * 2048u;
            unsigned ga = off + (unsigned)i * 2048u;
            asm volatile("global_load_async_to_lds_b128 %0, %1, %2"
                         :: "v"(la), "v"(ga), "s"(qg) : "memory");
        }
        asm volatile("s_wait_asynccnt 0" ::: "memory");
        __syncthreads();
    }

    // Q fragments: 4 chunks of K=32 over hd=128
    v16bf qf[4];
#pragma unroll
    for (int c = 0; c < 4; ++c)
#pragma unroll
        for (int e = 0; e < 16; ++e)
            qf[c][e] = bfu(sm.Qs[wave * 16 + l16][c * 32 + kmap(e, half)]);
    __syncthreads();   // compiler waits dscnt first -> safe to overwrite Qs

    v8f o[8];
#pragma unroll
    for (int i = 0; i < 8; ++i) o[i] = (v8f){0,0,0,0,0,0,0,0};
    float rm[8], rl[8];
#pragma unroll
    for (int e = 0; e < 8; ++e) { rm[e] = -INFINITY; rl[e] = 0.0f; }

    const int kend = qblk * 64 + 64;   // causal horizon (uniform per block)
    for (int kb = 0; kb < kend; kb += 32) {
        // ---- cooperative K/V tile staging: fp32 -> bf16 in LDS ----
#pragma unroll
        for (int i = 0; i < 8; ++i) {
            int idx = t + 128 * i;            // 0..1023
            int r   = idx >> 5;               // 0..31 (key row)
            int c   = (idx & 31) * 4;         // 0..124 (dim)
            const float4 kv4 = *(const float4*)(Kf + base + (size_t)(kb + r) * HEAD_DIM + c);
            const float4 vv4 = *(const float4*)(Vf + base + (size_t)(kb + r) * HEAD_DIM + c);
            *(unsigned*)&sm.kv.Ks[r][c]     = packbf(kv4.x, kv4.y);
            *(unsigned*)&sm.kv.Ks[r][c + 2] = packbf(kv4.z, kv4.w);
            *(unsigned*)&sm.kv.Vs[r][c]     = packbf(vv4.x, vv4.y);
            *(unsigned*)&sm.kv.Vs[r][c + 2] = packbf(vv4.z, vv4.w);
        }
        __syncthreads();

        // prefetch next K/V tile into cache while we compute this one
        if (kb + 32 < kend) {
            const size_t prow = base + (size_t)(kb + 32 + (t >> 2)) * HEAD_DIM + (t & 3) * 32;
            __builtin_prefetch(Kf + prow, 0, 1);
            __builtin_prefetch(Vf + prow, 0, 1);
        }

        // ---- S = Q K^T, two 16-key column tiles ----
        v8f s[2];
#pragma unroll
        for (int st = 0; st < 2; ++st) {
            s[st] = (v8f){0,0,0,0,0,0,0,0};
#pragma unroll
            for (int c = 0; c < 4; ++c) {
                v16bf kfrag;
#pragma unroll
                for (int e = 0; e < 16; ++e)
                    kfrag[e] = bfu(sm.kv.Ks[st * 16 + l16][c * 32 + kmap(e, half)]);
                s[st] = __builtin_amdgcn_wmma_f32_16x16x32_bf16(
                    false, qf[c], false, kfrag, (short)0, s[st], false, false);
            }
        }

        // ---- scale + causal mask ----
#pragma unroll
        for (int st = 0; st < 2; ++st)
#pragma unroll
            for (int e = 0; e < 8; ++e) {
                int key = kb + st * 16 + l16;
                int qr  = q0 + e + 8 * half;
                float v = s[st][e] * scale;
                s[st][e] = (key > qr) ? -INFINITY : v;
            }

        // ---- online softmax ----
        float alpha[8];
#pragma unroll
        for (int e = 0; e < 8; ++e) {
            float mx = red_max16(fmaxf(s[0][e], s[1][e]));
            float mn = fmaxf(rm[e], mx);
            alpha[e] = __expf(rm[e] - mn);
            rm[e] = mn;
            float p0 = __expf(s[0][e] - mn);
            float p1 = __expf(s[1][e] - mn);
            s[0][e] = p0; s[1][e] = p1;
            float rs = red_sum16(p0 + p1);
            rl[e] = rl[e] * alpha[e] + rs;
        }

        // rescale running O
#pragma unroll
        for (int nt = 0; nt < 8; ++nt)
#pragma unroll
            for (int e = 0; e < 8; ++e) o[nt][e] *= alpha[e];

        // ---- re-layout P (C-layout) -> A-operand via per-wave LDS tile ----
#pragma unroll
        for (int st = 0; st < 2; ++st)
#pragma unroll
            for (int e = 0; e < 8; ++e)
                sm.kv.Ps[wave][e + 8 * half][st * 16 + l16] = f2bf(s[st][e]);

        asm volatile("s_wait_dscnt 0" ::: "memory");

        v16bf pf;
#pragma unroll
        for (int e = 0; e < 16; ++e)
            pf[e] = bfu(sm.kv.Ps[wave][l16][kmap(e, half)]);

        // ---- O += P @ V ----
#pragma unroll
        for (int nt = 0; nt < 8; ++nt) {
            v16bf vfrag;
#pragma unroll
            for (int e = 0; e < 16; ++e)
                vfrag[e] = bfu(sm.kv.Vs[kmap(e, half)][nt * 16 + l16]);
            o[nt] = __builtin_amdgcn_wmma_f32_16x16x32_bf16(
                false, pf, false, vfrag, (short)0, o[nt], false, false);
        }
        __syncthreads();   // done with this K/V tile before restaging
    }

    // ---- epilogue: O / l -> [B,T,H*hd] ----
#pragma unroll
    for (int nt = 0; nt < 8; ++nt)
#pragma unroll
        for (int e = 0; e < 8; ++e) {
            int qr = q0 + e + 8 * half;
            int d  = nt * 16 + l16;
            Oout[((size_t)(b * SEQ + qr)) * MODEL_DIM + h * HEAD_DIM + d] =
                o[nt][e] / rl[e];
        }
}

// ---------------------------------------------------------------------------
// Launcher. d_in order: x, wq, bq, wk, bk, wv, bv, wo, bo (all fp32).
// d_out: [ out (B*T*D) | k (B*H*T*hd) | v (B*H*T*hd) ] fp32.
// ---------------------------------------------------------------------------
extern "C" void kernel_launch(void* const* d_in, const int* in_sizes, int n_in,
                              void* d_out, int out_size, void* d_ws, size_t ws_size,
                              hipStream_t stream)
{
    const float* x  = (const float*)d_in[0];
    const float* wq = (const float*)d_in[1];
    const float* bq = (const float*)d_in[2];
    const float* wk = (const float*)d_in[3];
    const float* bk = (const float*)d_in[4];
    const float* wv = (const float*)d_in[5];
    const float* bv = (const float*)d_in[6];
    const float* wo = (const float*)d_in[7];
    const float* bo = (const float*)d_in[8];

    const size_t outN = (size_t)BATCH * SEQ * MODEL_DIM;   // 16,777,216
    float* out  = (float*)d_out;
    float* Kout = out + outN;
    float* Vout = out + 2 * outN;

    // workspace: Q bf16 [B,H,T,hd] (32 MB) then attn output fp32 [B,T,D] (64 MB)
    unsigned short* Qws = (unsigned short*)d_ws;
    float* AttnWs = (float*)((char*)d_ws + (size_t)BATCH * NUM_HEADS * SEQ * HEAD_DIM * 2);

    dim3 ggrid(MROWS / TM, MODEL_DIM / TN);   // (64, 16)

    // QKV projections
    gemm_bias_kernel<<<ggrid, 256, 0, stream>>>(x, wq, bq, nullptr, Qws,
                                                MROWS, MODEL_DIM, MODEL_DIM, 2);
    gemm_bias_kernel<<<ggrid, 256, 0, stream>>>(x, wk, bk, Kout, nullptr,
                                                MROWS, MODEL_DIM, MODEL_DIM, 1);
    gemm_bias_kernel<<<ggrid, 256, 0, stream>>>(x, wv, bv, Vout, nullptr,
                                                MROWS, MODEL_DIM, MODEL_DIM, 1);

    // causal flash attention
    attn_kernel<<<dim3(BATCH * NUM_HEADS, SEQ / 64), 128, 0, stream>>>(
        Qws, Kout, Vout, AttnWs);

    // output projection
    gemm_bias_kernel<<<ggrid, 256, 0, stream>>>(AttnWs, wo, bo, out, nullptr,
                                                MROWS, MODEL_DIM, MODEL_DIM, 0);
}
